// ChainPOSAwareLM_35098472743431
// MI455X (gfx1250) — matrix operand order repaired and checked by reading, not compiled, not run on hardware
//
#include <hip/hip_runtime.h>
#include <hip/hip_bf16.h>
#include <cstdint>
#include <cstddef>

// ---------------------------------------------------------------------------
// CDNA5 (gfx1250) WMMA types and helpers.  All matrix math uses
// v_wmma_f32_16x16x32_f16 (f16 inputs, f32 accumulation), wave32, one wave
// per 16x16 output tile so EXEC is all-ones as the ISA requires.
// Fragment layouts follow ISA 7.12.2 exactly:
//   A (16x32, 16-bit): lane<16 holds row M=lane, elems 0..7 = K k0..k0+7,
//                      elems 8..15 = K k0+16..k0+23; lane>=16 shifts K by +8.
//   B (32x16, 16-bit): lane<16 holds col N, elems 0..7 = K k0..k0+7,
//                      elems 8..15 = K k0+16..k0+23; lane>=16 shifts K by +8 (same as A).
//   C (16x16, f32):    elem v -> row mbase + 8*(lane>=16) + v, col lane&15.
//
// All weights are pre-converted to f16 once per launch (each weight matrix is
// re-read 64x on the sequential critical path, so this removes all hot-loop
// v_cvt VALU work and halves per-step L2 weight traffic).
// ---------------------------------------------------------------------------

typedef __attribute__((ext_vector_type(16))) _Float16 v16h;
typedef __attribute__((ext_vector_type(8)))  float    v8f;

#define WMMA_F16(a, b, c) \
  __builtin_amdgcn_wmma_f32_16x16x32_f16(false, (a), false, (b), (short)0, (c), false, false)

__device__ __forceinline__ v16h load_a16(const _Float16* X, int ld, int row, int k0, int hi) {
  const _Float16* p = X + (size_t)row * ld + k0 + (hi ? 8 : 0);
  union { v16h v; uint4 q[2]; } u;
  u.q[0] = *(const uint4*)p;
  u.q[1] = *(const uint4*)(p + 16);
  return u.v;
}

__device__ __forceinline__ v16h load_b16(const _Float16* W, int ld, int n, int k0, int hi) {
  const _Float16* p = W + (size_t)n * ld + k0 + (hi ? 8 : 0);
  union { v16h v; uint4 q[2]; } u;
  u.q[0] = *(const uint4*)p;
  u.q[1] = *(const uint4*)(p + 16);
  return u.v;
}

__device__ __forceinline__ float sigmoidf_(float x) { return 1.0f / (1.0f + expf(-x)); }

// ---------------------------------------------------------------------------
// Generic Y = act(X @ W^T + b) tile kernel.  X f16 [M x K], W f16 [N x K],
// Y f16 [M x N].  grid = (N/16, M/16), block = 32 (one wave).
// ---------------------------------------------------------------------------
__global__ void __launch_bounds__(32)
gemm_act_kernel(const _Float16* X, int ldx, const _Float16* W, int ldw,
                const float* bias, int K, _Float16* Y, int ldy, int act) {
  const int lane = threadIdx.x, hi = lane >> 4, ln = lane & 15;
  const int nb = blockIdx.x * 16, mb = blockIdx.y * 16;
  const int rowA = mb + ln, n = nb + ln;
  v8f acc = {};
  for (int k0 = 0; k0 < K; k0 += 32) {
    v16h a = load_a16(X, ldx, rowA, k0, hi);
    v16h b = load_b16(W, ldw, n, k0, hi);
    acc = WMMA_F16(a, b, acc);
  }
  const float bv = bias ? bias[n] : 0.0f;
  const int rb = mb + hi * 8;
#pragma unroll
  for (int v = 0; v < 8; ++v) {
    float y = acc[v] + bv;
    if (act) y = tanhf(y);
    Y[(size_t)(rb + v) * ldy + n] = (_Float16)y;
  }
}

// ---------------------------------------------------------------------------
// Fused LSTM cell: gates = sum of up to 3 segment GEMMs (concatenated input),
// then the i/f/g/o pointwise update done entirely in C-fragment registers.
// Each wave owns one hidden tile and accumulates all four gate tiles so the
// A-fragment stream is reused 4x.  grid = (H/16, 2), block = 32.
// ---------------------------------------------------------------------------
__device__ __forceinline__ void seg_accum4(v8f* acc, const _Float16* X, int ldx,
                                           const _Float16* W, int ldw, int K,
                                           int rowA, int nb, int H, int hi) {
  for (int k0 = 0; k0 < K; k0 += 32) {
    if (k0 + 32 < K)
      __builtin_prefetch(X + (size_t)rowA * ldx + k0 + 32, 0, 3);
    v16h a = load_a16(X, ldx, rowA, k0, hi);
#pragma unroll
    for (int g = 0; g < 4; ++g) {
      v16h b = load_b16(W, ldw, nb + g * H, k0, hi);
      acc[g] = WMMA_F16(a, b, acc[g]);
    }
  }
}

__global__ void __launch_bounds__(32)
lstm_gate_kernel(const _Float16* X0, int ld0, const _Float16* W0, int lw0, int K0,
                 const _Float16* X1, int ld1, const _Float16* W1, int lw1, int K1,
                 const _Float16* X2, int ld2, const _Float16* W2, int lw2, int K2,
                 const float* bih, const float* bhh, int H,
                 float* c32, _Float16* hout) {
  const int lane = threadIdx.x, hi = lane >> 4, ln = lane & 15;
  const int j0 = blockIdx.x * 16, mb = blockIdx.y * 16;
  const int rowA = mb + ln, nb = j0 + ln;
  v8f acc[4] = {};
  if (K0) seg_accum4(acc, X0, ld0, W0, lw0, K0, rowA, nb, H, hi);
  if (K1) seg_accum4(acc, X1, ld1, W1, lw1, K1, rowA, nb, H, hi);
  if (K2) seg_accum4(acc, X2, ld2, W2, lw2, K2, rowA, nb, H, hi);
  const int j = j0 + ln;
  const float bi = bih[j]         + bhh[j];
  const float bf = bih[H + j]     + bhh[H + j];
  const float bg = bih[2 * H + j] + bhh[2 * H + j];
  const float bo = bih[3 * H + j] + bhh[3 * H + j];
  const int rb = mb + hi * 8;
#pragma unroll
  for (int v = 0; v < 8; ++v) {
    const int r = rb + v;
    const size_t idx = (size_t)r * H + j;
    float ig = sigmoidf_(acc[0][v] + bi);
    float fg = sigmoidf_(acc[1][v] + bf);
    float gg = tanhf(acc[2][v] + bg);
    float og = sigmoidf_(acc[3][v] + bo);
    float c = fg * c32[idx] + ig * gg;
    c32[idx] = c;
    hout[idx] = (_Float16)(og * tanhf(c));
  }
}

// ---------------------------------------------------------------------------
// Tiny POS head: 48 logits per batch row + log_softmax.  grid = B, block = 64.
// ---------------------------------------------------------------------------
__global__ void pos_logits_kernel(const _Float16* ph, const float* Wp,
                                  const float* bp, float* out) {
  const int b = blockIdx.x, tid = threadIdx.x;
  __shared__ float lg[48];
  __shared__ float m_s, l_s;
  if (tid < 48) {
    float s = bp[tid];
    const _Float16* x = ph + (size_t)b * 256;
    const float* w = Wp + (size_t)tid * 256;
    for (int k = 0; k < 256; ++k) s += (float)x[k] * w[k];
    lg[tid] = s;
  }
  __syncthreads();
  if (tid == 0) {
    float m = -3.0e38f;
    for (int i = 0; i < 48; ++i) m = fmaxf(m, lg[i]);
    float s = 0.0f;
    for (int i = 0; i < 48; ++i) s += expf(lg[i] - m);
    m_s = m; l_s = logf(s);
  }
  __syncthreads();
  if (tid < 48) out[(size_t)b * 48 + tid] = lg[tid] - m_s - l_s;
}

// ---------------------------------------------------------------------------
// Phase-2 vocab GEMM: logits[2048 x 32000] = E[2048 x 512] @ Wemb16^T + b2.
// 8 M-tiles per wave -> each B fragment is reused 8x.
// grid = (32000/16, 2048/128), block = 32.
// ---------------------------------------------------------------------------
#define MBLK 8
__global__ void __launch_bounds__(32)
vocab_gemm_kernel(const _Float16* E, const _Float16* Wt, const float* b2, float* out) {
  const int lane = threadIdx.x, hi = lane >> 4, ln = lane & 15;
  const int nb = blockIdx.x * 16;
  const int mg = blockIdx.y * (16 * MBLK);
  v8f acc[MBLK] = {};
  for (int k0 = 0; k0 < 512; k0 += 32) {
    if (k0 + 32 < 512)
      __builtin_prefetch(Wt + (size_t)(nb + ln) * 512 + k0 + 32 + (hi ? 16 : 0), 0, 3);
    v16h b = load_b16(Wt, 512, nb + ln, k0, hi);
#pragma unroll
    for (int m = 0; m < MBLK; ++m) {
      v16h a = load_a16(E, 512, mg + m * 16 + ln, k0, hi);
      acc[m] = WMMA_F16(a, b, acc[m]);
    }
  }
  const int n = nb + ln;
  const float bias = b2[n];
#pragma unroll
  for (int m = 0; m < MBLK; ++m) {
    const int rb = mg + m * 16 + hi * 8;
#pragma unroll
    for (int v = 0; v < 8; ++v)
      out[(size_t)(rb + v) * 32000 + n] = acc[m][v] + bias;
  }
}

// ---------------------------------------------------------------------------
// In-place row log_softmax over 32000 (online logsumexp, 2 passes).
// grid = 2048 rows, block = 256.
// ---------------------------------------------------------------------------
__global__ void logsoftmax_kernel(float* out) {
  const int row = blockIdx.x;
  float* p = out + (size_t)row * 32000;
  float m = -3.0e38f, s = 0.0f;
  for (int i = threadIdx.x; i < 32000; i += blockDim.x) {
    float x = p[i];
    if (x > m) { s = s * expf(m - x) + 1.0f; m = x; }
    else       { s += expf(x - m); }
  }
  __shared__ float sm[256], ss[256];
  sm[threadIdx.x] = m; ss[threadIdx.x] = s;
  __syncthreads();
  for (int off = blockDim.x / 2; off > 0; off >>= 1) {
    if (threadIdx.x < off) {
      float m2 = sm[threadIdx.x + off], s2 = ss[threadIdx.x + off];
      float M = fmaxf(sm[threadIdx.x], m2);
      ss[threadIdx.x] = ss[threadIdx.x] * expf(sm[threadIdx.x] - M) + s2 * expf(m2 - M);
      sm[threadIdx.x] = M;
    }
    __syncthreads();
  }
  const float M = sm[0], L = logf(ss[0]);
  for (int i = threadIdx.x; i < 32000; i += blockDim.x)
    p[i] = p[i] - M - L;
}

// ---------------------------------------------------------------------------
// Utility kernels.
// ---------------------------------------------------------------------------
__global__ void cvt_f32_f16_kernel(const float* src, _Float16* dst, size_t n) {
  size_t i = (size_t)blockIdx.x * blockDim.x + threadIdx.x;
  if (i < n) dst[i] = (_Float16)src[i];
}

__global__ void gather_embed_kernel(const int* idx, const float* table, int D, _Float16* out) {
  const int row = blockIdx.x;
  const int r = idx[row];
  for (int d = threadIdx.x; d < D; d += blockDim.x)
    out[(size_t)row * D + d] = (_Float16)table[(size_t)r * D + d];
}

__global__ void zero_u32_kernel(uint32_t* p, size_t n) {
  size_t i = (size_t)blockIdx.x * blockDim.x + threadIdx.x;
  if (i < n) p[i] = 0u;
}

// ---------------------------------------------------------------------------
// Host driver.
// ---------------------------------------------------------------------------
extern "C" void kernel_launch(void* const* d_in, const int* in_sizes, int n_in,
                              void* d_out, int out_size, void* d_ws, size_t ws_size,
                              hipStream_t stream) {
  (void)in_sizes; (void)n_in; (void)out_size; (void)ws_size;
  const int T = 64, B = 32;

  const int*   pos          = (const int*)d_in[0];
  const int*   word         = (const int*)d_in[1];
  const float* pos_emb_W    = (const float*)d_in[2];
  const float* word_emb_W   = (const float*)d_in[3];
  const float* w2p_W        = (const float*)d_in[4];
  const float* w2p_b        = (const float*)d_in[5];
  const float* p2w_W        = (const float*)d_in[6];
  const float* p2w_b        = (const float*)d_in[7];
  const float* p_Wih0       = (const float*)d_in[8];
  const float* p_Whh0       = (const float*)d_in[9];
  const float* p_bih0       = (const float*)d_in[10];
  const float* p_bhh0       = (const float*)d_in[11];
  const float* w_Wih0       = (const float*)d_in[12];
  const float* w_Whh0       = (const float*)d_in[13];
  const float* w_bih0       = (const float*)d_in[14];
  const float* w_bhh0       = (const float*)d_in[15];
  const float* w_Wih1       = (const float*)d_in[16];
  const float* w_Whh1       = (const float*)d_in[17];
  const float* w_bih1       = (const float*)d_in[18];
  const float* w_bhh1       = (const float*)d_in[19];
  const float* pos_proj_W   = (const float*)d_in[20];
  const float* pos_proj_b   = (const float*)d_in[21];
  const float* word_proj1_W = (const float*)d_in[22];
  const float* word_proj1_b = (const float*)d_in[23];
  const float* word_proj2_b = (const float*)d_in[24];

  float* outP = (float*)d_out;                 // [T,B,48]
  float* outW = outP + (size_t)T * B * 48;     // [T,B,32000]

  // ---- workspace partition (256B aligned regions) ----
  char* base = (char*)d_ws;
  size_t off = 0;
  auto take = [&](size_t bytes) -> char* {
    char* p = base + off;
    off = (off + bytes + 255) & ~(size_t)255;
    return p;
  };
  // f16 weight copies (converted once, re-read 64x from L2)
  _Float16* wemb16   = (_Float16*)take((size_t)32000 * 512 * 2);
  _Float16* w2p16    = (_Float16*)take((size_t)1024 * 1024 * 2);
  _Float16* p2w16    = (_Float16*)take((size_t)256 * 256 * 2);
  _Float16* pWih16   = (_Float16*)take((size_t)1024 * 1152 * 2);
  _Float16* pWhh16   = (_Float16*)take((size_t)1024 * 256 * 2);
  _Float16* wWih0_16 = (_Float16*)take((size_t)4096 * 768 * 2);
  _Float16* wWhh0_16 = (_Float16*)take((size_t)4096 * 1024 * 2);
  _Float16* wWih1_16 = (_Float16*)take((size_t)4096 * 1024 * 2);
  _Float16* wWhh1_16 = (_Float16*)take((size_t)4096 * 1024 * 2);
  _Float16* proj1_16 = (_Float16*)take((size_t)512 * 1024 * 2);
  // f16 activations
  _Float16* Pemb   = (_Float16*)take((size_t)T * B * 128 * 2);
  _Float16* Wemb   = (_Float16*)take((size_t)T * B * 512 * 2);
  _Float16* E16    = (_Float16*)take((size_t)T * B * 512 * 2);
  _Float16* lastw  = (_Float16*)take((size_t)B * 1024 * 2);
  _Float16* lastp  = (_Float16*)take((size_t)B * 256 * 2);
  char* state = base + off;                                     // zeroed each call
  _Float16* ph[2]  = { (_Float16*)take((size_t)B * 256 * 2),  (_Float16*)take((size_t)B * 256 * 2) };
  _Float16* wh0[2] = { (_Float16*)take((size_t)B * 1024 * 2), (_Float16*)take((size_t)B * 1024 * 2) };
  _Float16* wh1[2] = { (_Float16*)take((size_t)B * 1024 * 2), (_Float16*)take((size_t)B * 1024 * 2) };
  float* pc  = (float*)take((size_t)B * 256 * 4);
  float* wc0 = (float*)take((size_t)B * 1024 * 4);
  float* wc1 = (float*)take((size_t)B * 1024 * 4);
  const size_t state_bytes = (size_t)((base + off) - state);

  // ---- prologue: weight conversions, gathers, state zeroing ----
  auto cvt = [&](const float* src, _Float16* dst, size_t n) {
    cvt_f32_f16_kernel<<<dim3((unsigned)((n + 255) / 256)), 256, 0, stream>>>(src, dst, n);
  };
  cvt(word_emb_W,   wemb16,   (size_t)32000 * 512);
  cvt(w2p_W,        w2p16,    (size_t)1024 * 1024);
  cvt(p2w_W,        p2w16,    (size_t)256 * 256);
  cvt(p_Wih0,       pWih16,   (size_t)1024 * 1152);
  cvt(p_Whh0,       pWhh16,   (size_t)1024 * 256);
  cvt(w_Wih0,       wWih0_16, (size_t)4096 * 768);
  cvt(w_Whh0,       wWhh0_16, (size_t)4096 * 1024);
  cvt(w_Wih1,       wWih1_16, (size_t)4096 * 1024);
  cvt(w_Whh1,       wWhh1_16, (size_t)4096 * 1024);
  cvt(word_proj1_W, proj1_16, (size_t)512 * 1024);

  gather_embed_kernel<<<T * B, 128, 0, stream>>>(pos,  pos_emb_W,  128, Pemb);
  gather_embed_kernel<<<T * B, 128, 0, stream>>>(word, word_emb_W, 512, Wemb);
  {
    size_t n = state_bytes / 4;
    zero_u32_kernel<<<dim3((unsigned)((n + 255) / 256)), 256, 0, stream>>>((uint32_t*)state, n);
  }

  // ---- phase 1: sequential recurrence (double-buffered hidden states) ----
  for (int t = 0; t < T; ++t) {
    const int pr = t & 1, nx = pr ^ 1;

    // last_w = tanh(wh1_prev @ w2p_W^T + b)                          [32,1024]
    gemm_act_kernel<<<dim3(64, 2), 32, 0, stream>>>(
        wh1[pr], 1024, w2p16, 1024, w2p_b, 1024, lastw, 1024, 1);

    // POS-LSTM: x = [p_emb_t | last_w], h = ph_prev  -> ph_new, pc   [32,256]
    lstm_gate_kernel<<<dim3(16, 2), 32, 0, stream>>>(
        Pemb + (size_t)t * B * 128, 128, pWih16,       1152, 128,
        lastw,                     1024, pWih16 + 128, 1152, 1024,
        ph[pr],                     256, pWhh16,        256, 256,
        p_bih0, p_bhh0, 256, pc, ph[nx]);

    // last_p = tanh(ph_new @ p2w_W^T + b)                            [32,256]
    gemm_act_kernel<<<dim3(16, 2), 32, 0, stream>>>(
        ph[nx], 256, p2w16, 256, p2w_b, 256, lastp, 256, 1);

    // POS head: log_softmax(ph_new @ pos_proj^T + b)                 [32,48]
    pos_logits_kernel<<<B, 64, 0, stream>>>(
        ph[nx], pos_proj_W, pos_proj_b, outP + (size_t)t * B * 48);

    // word-LSTM layer 0: x = [w_emb_t | last_p], h = wh0_prev        [32,1024]
    lstm_gate_kernel<<<dim3(64, 2), 32, 0, stream>>>(
        Wemb + (size_t)t * B * 512, 512, wWih0_16,       768, 512,
        lastp,                      256, wWih0_16 + 512, 768, 256,
        wh0[pr],                   1024, wWhh0_16,      1024, 1024,
        w_bih0, w_bhh0, 1024, wc0, wh0[nx]);

    // word-LSTM layer 1: x = wh0_new, h = wh1_prev                   [32,1024]
    lstm_gate_kernel<<<dim3(64, 2), 32, 0, stream>>>(
        wh0[nx], 1024, wWih1_16, 1024, 1024,
        wh1[pr], 1024, wWhh1_16, 1024, 1024,
        (const _Float16*)nullptr, 0, (const _Float16*)nullptr, 0, 0,
        w_bih1, w_bhh1, 1024, wc1, wh1[nx]);

    // E[t] = wh1_new @ word_proj1^T + b                              [32,512]
    gemm_act_kernel<<<dim3(32, 2), 32, 0, stream>>>(
        wh1[nx], 1024, proj1_16, 1024, word_proj1_b, 1024,
        E16 + (size_t)t * B * 512, 512, 0);
  }

  // ---- phase 2: tied vocab projection + log_softmax ----
  vocab_gemm_kernel<<<dim3(32000 / 16, (T * B) / (16 * MBLK)), 32, 0, stream>>>(
      E16, wemb16, word_proj2_b, outW);
  logsoftmax_kernel<<<T * B, 256, 0, stream>>>(outW);
}
